// TopK_10307921510449
// MI455X (gfx1250) — matrix operand order repaired and checked by reading, not compiled, not run on hardware
//
#include <hip/hip_runtime.h>
#include <hip/hip_bf16.h>
#include <math.h>

typedef __attribute__((ext_vector_type(16))) _Float16 v16h;
typedef __attribute__((ext_vector_type(8)))  float    v8f;

#define N_NODES 100000
#define FEATS   1024
#define KSEL    4096
#define NTILES  (N_NODES / 16)   // 6250 exact
#define HBINS   2048

// ---- workspace layout (bytes) ----
#define WS_SCORES   0            // N_NODES floats          (400000 B)
#define WS_HIST     400384       // 2048 u32                (8192 B)
#define WS_STATE    408576       // 8 u32: [0]=remaining [1]=b0 [2]=b1 [3]=T
#define WS_COUNTER  408608       // 1 u32
#define WS_CAND     408616       // 4096 u64 (8-aligned)    (32768 B)
#define WS_SELIDX   441384       // 4096 u32
#define WS_TVAL     457768       // 4096 f32
#define WS_INVNORM  474152       // 1 f32

// monotonic float -> uint key (order-preserving)
__device__ __forceinline__ unsigned score_key(float f) {
    unsigned u = __float_as_uint(f);
    return (u & 0x80000000u) ? ~u : (u | 0x80000000u);
}

// ---------------- prep: ||w||, zero hist/counter, init select state --------
__global__ void prep_kernel(const float* __restrict__ w, float* inv_norm,
                            unsigned* hist, unsigned* state, unsigned* counter) {
    __shared__ float red[256];
    float s = 0.f;
    for (int i = threadIdx.x; i < FEATS; i += 256) { float v = w[i]; s += v * v; }
    red[threadIdx.x] = s;
    __syncthreads();
    for (int off = 128; off > 0; off >>= 1) {
        if (threadIdx.x < off) red[threadIdx.x] += red[threadIdx.x + off];
        __syncthreads();
    }
    if (threadIdx.x == 0) {
        *inv_norm = rsqrtf(red[0]);
        *counter  = 0u;
        state[0] = KSEL; state[1] = 0u; state[2] = 0u; state[3] = 0u;
    }
    for (int i = threadIdx.x; i < HBINS; i += 256) hist[i] = 0u;
}

// ---------------- GEMV via WMMA, hi/lo f16 error compensation --------------
// Each wave: one 16-row tile. B = w broadcast to all 16 columns, so every
// column of D holds the 16 dot products. 3 WMMAs per 32-wide K chunk.
__global__ void score_wmma_kernel(const float* __restrict__ X,
                                  const float* __restrict__ mask,
                                  const float* __restrict__ w,
                                  float* __restrict__ scores) {
    __shared__ _Float16 wh[FEATS];
    __shared__ _Float16 wl[FEATS];
    for (int i = threadIdx.x; i < FEATS; i += 256) {
        float v = w[i];
        _Float16 h = (_Float16)v;
        wh[i] = h;
        wl[i] = (_Float16)(v - (float)h);
    }
    __syncthreads();

    int wave = threadIdx.x >> 5;
    int lane = threadIdx.x & 31;
    int tile = blockIdx.x * 8 + wave;
    if (tile >= NTILES) return;              // uniform per wave -> EXEC all-1s

    int row_base = tile * 16;
    int m  = lane & 15;
    int kg = (lane >> 4) << 3;               // A lane-group K offset: 0 / 8
    int kb = (lane >> 4) << 4;               // unused: B is read with A's k order (kw = ka)
    const float* xrow = X + (size_t)(row_base + m) * FEATS;

    v8f acc = {};
    for (int c = 0; c < FEATS; c += 32) {
        v16h ah, al, bh, bl;
#pragma unroll
        for (int v = 0; v < 8; ++v) {
            int ka = (v < 4) ? (kg + 2 * v) : (16 + kg + 2 * (v - 4));
            float2 xv = *(const float2*)(xrow + c + ka);
            _Float16 h0 = (_Float16)xv.x;
            _Float16 h1 = (_Float16)xv.y;
            ah[2 * v]     = h0;
            ah[2 * v + 1] = h1;
            al[2 * v]     = (_Float16)(xv.x - (float)h0);
            al[2 * v + 1] = (_Float16)(xv.y - (float)h1);
            int kw = ka;
            bh[2 * v]     = wh[c + kw];
            bh[2 * v + 1] = wh[c + kw + 1];
            bl[2 * v]     = wl[c + kw];
            bl[2 * v + 1] = wl[c + kw + 1];
        }
        acc = __builtin_amdgcn_wmma_f32_16x16x32_f16(false, al, false, bh, (short)0, acc, false, false);
        acc = __builtin_amdgcn_wmma_f32_16x16x32_f16(false, ah, false, bl, (short)0, acc, false, false);
        acc = __builtin_amdgcn_wmma_f32_16x16x32_f16(false, ah, false, bh, (short)0, acc, false, false);
    }
    // D layout: lanes 0-15 VGPR r -> M=r; lanes 16-31 VGPR r -> M=8+r.
    // All columns identical (broadcast B): lane 0 / lane 16 emit the scores.
    if (lane == 0) {
#pragma unroll
        for (int r = 0; r < 8; ++r)
            scores[row_base + r] = acc[r] + mask[row_base + r];
    } else if (lane == 16) {
#pragma unroll
        for (int r = 0; r < 8; ++r)
            scores[row_base + 8 + r] = acc[r] + mask[row_base + 8 + r];
    }
}

// ---------------- radix-select histograms (11 | 11 | 10 bits) --------------
__global__ void hist_kernel(const float* __restrict__ scores, unsigned* hist,
                            const unsigned* __restrict__ state, int level) {
    __shared__ unsigned lh[HBINS];
    for (int i = threadIdx.x; i < HBINS; i += 256) lh[i] = 0u;
    __syncthreads();
    int i = blockIdx.x * 256 + threadIdx.x;
    if (i < N_NODES) {
        unsigned key = score_key(scores[i]);
        if (level == 0) {
            atomicAdd(&lh[key >> 21], 1u);
        } else if (level == 1) {
            if ((key >> 21) == state[1]) atomicAdd(&lh[(key >> 10) & 2047u], 1u);
        } else {
            if ((key >> 10) == ((state[1] << 11) | state[2])) atomicAdd(&lh[key & 1023u], 1u);
        }
    }
    __syncthreads();
    for (int b = threadIdx.x; b < HBINS; b += 256) {
        unsigned c = lh[b];
        if (c) atomicAdd(&hist[b], c);
    }
}

// top-down scan: locate bin holding the remaining-th largest; re-zero hist
__global__ void scan_kernel(unsigned* hist, unsigned* state, int level) {
    if (threadIdx.x == 0) {
        int bins = (level == 2) ? 1024 : HBINS;
        unsigned remaining = state[0];
        unsigned cum = 0;
        int sel = 0;
        for (int b = bins - 1; b >= 0; --b) {
            unsigned c = hist[b];
            if (cum + c >= remaining) { sel = b; break; }
            cum += c;
        }
        state[0] = remaining - cum;
        if (level == 0)      state[1] = (unsigned)sel;
        else if (level == 1) state[2] = (unsigned)sel;
        else                 state[3] = (state[1] << 21) | (state[2] << 10) | (unsigned)sel;
    }
    __syncthreads();
    for (int i = threadIdx.x; i < HBINS; i += 256) hist[i] = 0u;
}

// ---------------- collect key >= T -----------------------------------------
__global__ void collect_kernel(const float* __restrict__ scores,
                               const unsigned* __restrict__ state,
                               unsigned* counter, unsigned long long* cand) {
    int i = blockIdx.x * 256 + threadIdx.x;
    if (i >= N_NODES) return;
    unsigned key = score_key(scores[i]);
    if (key >= state[3]) {
        unsigned pos = atomicAdd(counter, 1u);
        // pack so ascending sort == (value desc, index asc) like lax.top_k
        if (pos < KSEL)
            cand[pos] = ((unsigned long long)(~key) << 32) | (unsigned)i;
    }
}

// ---------------- bitonic sort 4096 in LDS, emit idx + tanh ----------------
__global__ void sort_emit_kernel(const unsigned long long* __restrict__ cand,
                                 const unsigned* __restrict__ counter,
                                 const float* __restrict__ scores,
                                 const float* __restrict__ inv_norm,
                                 unsigned* __restrict__ sel_idx,
                                 float* __restrict__ tval) {
    __shared__ unsigned long long s[KSEL];          // 32 KB
    unsigned cnt = *counter; if (cnt > KSEL) cnt = KSEL;
    for (int i = threadIdx.x; i < KSEL; i += 1024)
        s[i] = (i < (int)cnt) ? cand[i] : 0xFFFFFFFFFFFFFFFFull;
    __syncthreads();
    for (int k = 2; k <= KSEL; k <<= 1) {
        for (int j = k >> 1; j > 0; j >>= 1) {
            for (int i = threadIdx.x; i < KSEL; i += 1024) {
                int l = i ^ j;
                if (l > i) {
                    unsigned long long a = s[i], b = s[l];
                    bool asc = ((i & k) == 0);
                    if ((a > b) == asc) { s[i] = b; s[l] = a; }
                }
            }
            __syncthreads();
        }
    }
    float inv = *inv_norm;
    for (int i = threadIdx.x; i < KSEL; i += 1024) {
        unsigned idx = (unsigned)(s[i] & 0xFFFFFFFFu);
        if (idx >= N_NODES) idx = N_NODES - 1;       // pad-with-last safety
        sel_idx[i] = idx;
        tval[i] = tanhf(scores[idx] * inv);
    }
}

// ---------------- gather: out[f*K + k] = X[idx[k], f] * tanh --------------
__global__ void gather_kernel(const float* __restrict__ X,
                              const unsigned* __restrict__ sel_idx,
                              const float* __restrict__ tval,
                              float* __restrict__ out) {
    int gid = blockIdx.x * 256 + threadIdx.x;
    int k = gid & (KSEL - 1);
    int f = gid >> 12;                                // KSEL == 2^12
    out[gid] = X[(size_t)sel_idx[k] * FEATS + f] * tval[k];
}

extern "C" void kernel_launch(void* const* d_in, const int* in_sizes, int n_in,
                              void* d_out, int out_size, void* d_ws, size_t ws_size,
                              hipStream_t stream) {
    (void)in_sizes; (void)n_in; (void)out_size; (void)ws_size;
    const float* X    = (const float*)d_in[0];   // [N, FEATS]
    const float* mask = (const float*)d_in[1];   // [N, 1]
    const float* w    = (const float*)d_in[2];   // [FEATS, 1]
    float* out = (float*)d_out;                  // [FEATS, KSEL]

    char* ws = (char*)d_ws;
    float*              scores  = (float*)(ws + WS_SCORES);
    unsigned*           hist    = (unsigned*)(ws + WS_HIST);
    unsigned*           state   = (unsigned*)(ws + WS_STATE);
    unsigned*           counter = (unsigned*)(ws + WS_COUNTER);
    unsigned long long* cand    = (unsigned long long*)(ws + WS_CAND);
    unsigned*           sel_idx = (unsigned*)(ws + WS_SELIDX);
    float*              tval    = (float*)(ws + WS_TVAL);
    float*              invn    = (float*)(ws + WS_INVNORM);

    const int nb = (N_NODES + 255) / 256;

    prep_kernel<<<1, 256, 0, stream>>>(w, invn, hist, state, counter);
    score_wmma_kernel<<<(NTILES + 7) / 8, 256, 0, stream>>>(X, mask, w, scores);

    hist_kernel<<<nb, 256, 0, stream>>>(scores, hist, state, 0);
    scan_kernel<<<1, 256, 0, stream>>>(hist, state, 0);
    hist_kernel<<<nb, 256, 0, stream>>>(scores, hist, state, 1);
    scan_kernel<<<1, 256, 0, stream>>>(hist, state, 1);
    hist_kernel<<<nb, 256, 0, stream>>>(scores, hist, state, 2);
    scan_kernel<<<1, 256, 0, stream>>>(hist, state, 2);

    collect_kernel<<<nb, 256, 0, stream>>>(scores, state, counter, cand);
    sort_emit_kernel<<<1, 1024, 0, stream>>>(cand, counter, scores, invn, sel_idx, tval);
    gather_kernel<<<(FEATS * KSEL) / 256, 256, 0, stream>>>(X, sel_idx, tval, out);
}
